// InternLM2Attention_1726576856922
// MI455X (gfx1250) — matrix operand order repaired and checked by reading, compiled, not run on hardware
//
#include <hip/hip_runtime.h>
#include <hip/hip_bf16.h>
#include <stdint.h>

// ---------------- problem constants ----------------
#define SEQ   2048
#define HID   4096
#define NH    32
#define NKV   8
#define GRP   4
#define HD    128
#define QKV_OUT 6144          // (NH + 2*NKV) * HD
#define LORA_R  256

typedef __attribute__((ext_vector_type(16))) __bf16 v16bf;
typedef __attribute__((ext_vector_type(8)))  float  v8f;

union FragAB {            // 16 bf16 = one WMMA A or B operand (wave32)
    v16bf v;
    uint4 u4[2];
};

static __device__ __forceinline__ __bf16 f2bf(float f) { return (__bf16)f; }

// =====================================================================
// GEMM: C[M,N] = X[M,K] @ W[N,K]^T   (bf16 WMMA, f32 accumulate)
// ADDMODE==1: C[m,n] += mask[m] * acc   (LoRA add, scaling == 1.0)
// M%128==0, N%128==0, K%32==0.  Double-buffered LDS, 1 barrier / K-step.
// =====================================================================
#define BM 128
#define BN 128
#define BK 32
#define LSTR 40               // LDS row stride in bf16 elements (pad vs 32)

template <int ADDMODE>
__global__ __launch_bounds__(256)
void gemm_bf16_wmma(const float* __restrict__ X, const float* __restrict__ W,
                    float* __restrict__ C, int M, int N, int K,
                    const unsigned char* __restrict__ mask)
{
    __shared__ __align__(16) __bf16 Xs[2][BM * LSTR];
    __shared__ __align__(16) __bf16 Ws[2][BN * LSTR];

    const int tid  = threadIdx.x;
    const int lane = tid & 31;
    const int wid  = tid >> 5;
    const int ln   = lane & 15;     // 0..15
    const int lh   = lane >> 4;     // half 0/1
    const int wave_m = wid & 3;     // 4 waves over M -> 32 rows each
    const int wave_n = wid >> 2;    // 2 waves over N -> 64 cols each
    const int blockM = blockIdx.y * BM;
    const int blockN = blockIdx.x * BN;

    // per-thread staging coordinates (4 float4 of X and W per thread)
    int rr[4], gg[4];
#pragma unroll
    for (int i = 0; i < 4; ++i) {
        int idx = tid + i * 256;          // 0..1023
        rr[i] = idx >> 3;                 // tile row 0..127
        gg[i] = idx & 7;                  // 4-float group 0..7
    }

    v8f acc[2][4];
#pragma unroll
    for (int i = 0; i < 2; ++i)
#pragma unroll
        for (int j = 0; j < 4; ++j) acc[i][j] = (v8f)0.0f;

    float4 xr[4], wr[4];
    // ---- preload tile 0: global -> regs -> LDS[0]
#pragma unroll
    for (int i = 0; i < 4; ++i) {
        xr[i] = *(const float4*)(X + (size_t)(blockM + rr[i]) * K + gg[i] * 4);
        wr[i] = *(const float4*)(W + (size_t)(blockN + rr[i]) * K + gg[i] * 4);
    }
#pragma unroll
    for (int i = 0; i < 4; ++i) {
        __bf16* dx = &Xs[0][rr[i] * LSTR + gg[i] * 4];
        dx[0] = f2bf(xr[i].x); dx[1] = f2bf(xr[i].y); dx[2] = f2bf(xr[i].z); dx[3] = f2bf(xr[i].w);
        __bf16* dw = &Ws[0][rr[i] * LSTR + gg[i] * 4];
        dw[0] = f2bf(wr[i].x); dw[1] = f2bf(wr[i].y); dw[2] = f2bf(wr[i].z); dw[3] = f2bf(wr[i].w);
    }
    __syncthreads();

    const int nk = K / BK;
    for (int kt = 0; kt < nk; ++kt) {
        const int cur = kt & 1;
        const int k1  = (kt + 1) * BK;

        // ---- prefetch tile kt+1 into registers (overlaps with WMMAs below)
        if (kt + 1 < nk) {
#pragma unroll
            for (int i = 0; i < 4; ++i) {
                xr[i] = *(const float4*)(X + (size_t)(blockM + rr[i]) * K + k1 + gg[i] * 4);
                wr[i] = *(const float4*)(W + (size_t)(blockN + rr[i]) * K + k1 + gg[i] * 4);
            }
            if (kt + 2 < nk) {   // warm L2 for the tile after that (global_prefetch_b8)
                __builtin_prefetch(X + (size_t)(blockM + rr[0]) * K + k1 + BK, 0, 1);
                __builtin_prefetch(W + (size_t)(blockN + rr[0]) * K + k1 + BK, 0, 1);
            }
        }

        // ---- fragments from LDS[cur] + 8 WMMAs
        FragAB a[2], b[4];
#pragma unroll
        for (int mt = 0; mt < 2; ++mt) {
            int m = wave_m * 32 + mt * 16 + ln;
            a[mt].u4[0] = *(const uint4*)&Xs[cur][m * LSTR + lh * 8];
            a[mt].u4[1] = *(const uint4*)&Xs[cur][m * LSTR + 16 + lh * 8];
        }
#pragma unroll
        for (int nt = 0; nt < 4; ++nt) {
            int n = wave_n * 64 + nt * 16 + ln;
            b[nt].u4[0] = *(const uint4*)&Ws[cur][n * LSTR + lh * 8];
            b[nt].u4[1] = *(const uint4*)&Ws[cur][n * LSTR + 16 + lh * 8];
        }
#pragma unroll
        for (int mt = 0; mt < 2; ++mt)
#pragma unroll
            for (int nt = 0; nt < 4; ++nt)
                acc[mt][nt] = __builtin_amdgcn_wmma_f32_16x16x32_bf16(
                    false, a[mt].v, false, b[nt].v, (short)0, acc[mt][nt], false, false);

        // ---- store prefetched tile into the other buffer, single barrier
        if (kt + 1 < nk) {
#pragma unroll
            for (int i = 0; i < 4; ++i) {
                __bf16* dx = &Xs[1 - cur][rr[i] * LSTR + gg[i] * 4];
                dx[0] = f2bf(xr[i].x); dx[1] = f2bf(xr[i].y); dx[2] = f2bf(xr[i].z); dx[3] = f2bf(xr[i].w);
                __bf16* dw = &Ws[1 - cur][rr[i] * LSTR + gg[i] * 4];
                dw[0] = f2bf(wr[i].x); dw[1] = f2bf(wr[i].y); dw[2] = f2bf(wr[i].z); dw[3] = f2bf(wr[i].w);
            }
        }
        __syncthreads();
    }

    // ---- epilogue: D layout -> lane holds N=ln, M = r + 8*lh
    if (ADDMODE) {
        float mkv[2][8];
#pragma unroll
        for (int mt = 0; mt < 2; ++mt)
#pragma unroll
            for (int r = 0; r < 8; ++r) {
                int m = blockM + wave_m * 32 + mt * 16 + r + 8 * lh;
                mkv[mt][r] = mask[m] ? 1.0f : 0.0f;
            }
#pragma unroll
        for (int mt = 0; mt < 2; ++mt)
#pragma unroll
            for (int nt = 0; nt < 4; ++nt)
#pragma unroll
                for (int r = 0; r < 8; ++r) {
                    int m = blockM + wave_m * 32 + mt * 16 + r + 8 * lh;
                    int n = blockN + wave_n * 64 + nt * 16 + ln;
                    size_t o = (size_t)m * N + n;
                    C[o] = C[o] + mkv[mt][r] * acc[mt][nt][r];
                }
    } else {
#pragma unroll
        for (int mt = 0; mt < 2; ++mt)
#pragma unroll
            for (int nt = 0; nt < 4; ++nt)
#pragma unroll
                for (int r = 0; r < 8; ++r) {
                    int m = blockM + wave_m * 32 + mt * 16 + r + 8 * lh;
                    int n = blockN + wave_n * 64 + nt * 16 + ln;
                    C[(size_t)m * N + n] = acc[mt][nt][r];
                }
    }
}

// =====================================================================
// RoPE + repack: qkv f32 [S][NKV][6][HD]  ->
//   Qb  bf16 [NH][S][HD]   (RoPE'd, pre-scaled by 1/sqrt(HD))
//   Kb  bf16 [NKV][S][HD]  (row-major == B-frag layout for Q@K^T)
//   Vt  bf16 [NKV][HD][S]  (transposed == B-frag layout for P@V)
// =====================================================================
__global__ __launch_bounds__(256)
void rope_pack(const float* __restrict__ qkv, const int* __restrict__ pos,
               __bf16* __restrict__ Qb, __bf16* __restrict__ Kb,
               __bf16* __restrict__ Vt)
{
    int idx  = blockIdx.x * 256 + threadIdx.x;   // < 2048*8*6*64
    int d    = idx & 63;
    int slot = (idx >> 6) % 6;
    int kvh  = ((idx >> 6) / 6) & 7;
    int s    = idx / (64 * 6 * 8);
    const float* src = qkv + (size_t)s * QKV_OUT + (size_t)(kvh * 6 + slot) * HD;

    if (slot == 5) {                      // V: transpose only
        float v0 = src[d], v1 = src[d + 64];
        Vt[((size_t)kvh * HD + d) * SEQ + s]      = f2bf(v0);
        Vt[((size_t)kvh * HD + d + 64) * SEQ + s] = f2bf(v1);
        return;
    }
    float p   = (float)pos[s];
    // inv_freq = 10000^(-2d/128); ln(10000) = 9.210340371976184
    float inv = __expf(-(float)(2 * d) * (1.0f / 128.0f) * 9.210340371976184f);
    float ang = p * inv, c, sn;
    __sincosf(ang, &c, &sn);
    float x1 = src[d], x2 = src[d + 64];
    float o1 = x1 * c - x2 * sn;
    float o2 = x2 * c + x1 * sn;
    if (slot == 4) {                      // K
        __bf16* dst = Kb + ((size_t)kvh * SEQ + s) * HD;
        dst[d] = f2bf(o1); dst[d + 64] = f2bf(o2);
    } else {                              // Q head = kvh*4 + slot, fold 1/sqrt(128)
        const float qs = 0.08838834764831845f;
        int h = kvh * GRP + slot;
        __bf16* dst = Qb + ((size_t)h * SEQ + s) * HD;
        dst[d] = f2bf(o1 * qs); dst[d + 64] = f2bf(o2 * qs);
    }
}

// =====================================================================
// Flash attention (causal, GQA 4:1). One wave = 16 queries of one head.
// Block = 8 waves (independent q tiles). Per 32-wide KV block:
//   8 score WMMAs, online softmax, P->LDS->A-frag, 8 P@V WMMAs.
// =====================================================================
__global__ __launch_bounds__(256)
void attn_wmma(const __bf16* __restrict__ Qb, const __bf16* __restrict__ Kb,
               const __bf16* __restrict__ Vt, float* __restrict__ Out)
{
    __shared__ __align__(16) __bf16 Pst[8][16 * LSTR];   // per-wave 16x32 P tile

    const int tid  = threadIdx.x;
    const int lane = tid & 31;
    const int wid  = tid >> 5;
    const int ln   = lane & 15;
    const int lh   = lane >> 4;
    const int h    = blockIdx.y;
    const int kvh  = h >> 2;
    const int q0   = (blockIdx.x * 8 + wid) * 16;
    __bf16* P = &Pst[wid][0];

    // Q fragments: 4 chunks of K-dim 32, held in registers for the whole row
    FragAB qf[4];
    const __bf16* Qrow = Qb + ((size_t)h * SEQ + q0) * HD;
#pragma unroll
    for (int c = 0; c < 4; ++c) {
        const __bf16* r0 = Qrow + (size_t)ln * HD + c * 32;
        qf[c].u4[0] = *(const uint4*)(r0 + lh * 8);
        qf[c].u4[1] = *(const uint4*)(r0 + 16 + lh * 8);
    }

    v8f o[8];
#pragma unroll
    for (int dt = 0; dt < 8; ++dt) o[dt] = (v8f)0.0f;
    float m_[8], l_[8];
#pragma unroll
    for (int r = 0; r < 8; ++r) { m_[r] = -1e30f; l_[r] = 0.0f; }

    const int nb = (q0 + 16 + 31) / 32;   // causal: only blocks touching kv <= q0+15
    for (int blk = 0; blk < nb; ++blk) {
        const int c0 = blk * 32;

        // warm L2 for the next KV block while we compute this one
        if (blk + 1 < nb) {
            __builtin_prefetch(Kb + ((size_t)kvh * SEQ + c0 + 32 + ln) * HD, 0, 1);
            __builtin_prefetch(Vt + ((size_t)kvh * HD + ln) * SEQ + c0 + 32, 0, 1);
        }

        // ---- scores: two 16-col tiles, K-dim 128 in 4 WMMAs each
        v8f s0 = (v8f)0.0f, s1 = (v8f)0.0f;
        {
            const __bf16* Kr0 = Kb + ((size_t)kvh * SEQ + c0 + ln) * HD;
            const __bf16* Kr1 = Kr0 + (size_t)16 * HD;
#pragma unroll
            for (int c = 0; c < 4; ++c) {
                FragAB kf0, kf1;
                kf0.u4[0] = *(const uint4*)(Kr0 + c * 32 + lh * 8);
                kf0.u4[1] = *(const uint4*)(Kr0 + c * 32 + 16 + lh * 8);
                kf1.u4[0] = *(const uint4*)(Kr1 + c * 32 + lh * 8);
                kf1.u4[1] = *(const uint4*)(Kr1 + c * 32 + 16 + lh * 8);
                s0 = __builtin_amdgcn_wmma_f32_16x16x32_bf16(false, qf[c].v, false, kf0.v,
                                                             (short)0, s0, false, false);
                s1 = __builtin_amdgcn_wmma_f32_16x16x32_bf16(false, qf[c].v, false, kf1.v,
                                                             (short)0, s1, false, false);
            }
        }

        // ---- causal mask + online softmax (rows live in 16-lane halves)
#pragma unroll
        for (int r = 0; r < 8; ++r) {
            int qrow = q0 + r + 8 * lh;
            float v0 = (c0 + ln      <= qrow) ? s0[r] : -1e30f;
            float v1 = (c0 + 16 + ln <= qrow) ? s1[r] : -1e30f;
            float mx = fmaxf(v0, v1);
#pragma unroll
            for (int d = 1; d < 16; d <<= 1) mx = fmaxf(mx, __shfl_xor(mx, d));
            float mnew  = fmaxf(m_[r], mx);
            float scale = __expf(m_[r] - mnew);
            float p0 = __expf(v0 - mnew);
            float p1 = __expf(v1 - mnew);
            float rs = p0 + p1;
#pragma unroll
            for (int d = 1; d < 16; d <<= 1) rs += __shfl_xor(rs, d);
            l_[r] = l_[r] * scale + rs;
            m_[r] = mnew;
#pragma unroll
            for (int dt = 0; dt < 8; ++dt) o[dt][r] *= scale;
            // stash P (bf16) into per-wave LDS tile, row = q-row, col = kv-local
            P[(r + 8 * lh) * LSTR + ln]      = f2bf(p0);
            P[(r + 8 * lh) * LSTR + 16 + ln] = f2bf(p1);
        }

        // ---- re-load P as an A fragment (16x32) and do P @ V
        FragAB pa;
        pa.u4[0] = *(const uint4*)&P[ln * LSTR + lh * 8];
        pa.u4[1] = *(const uint4*)&P[ln * LSTR + 16 + lh * 8];
#pragma unroll
        for (int dt = 0; dt < 8; ++dt) {
            FragAB vf;
            const __bf16* Vr = Vt + ((size_t)kvh * HD + dt * 16 + ln) * SEQ + c0;
            vf.u4[0] = *(const uint4*)(Vr + lh * 8);
            vf.u4[1] = *(const uint4*)(Vr + 16 + lh * 8);
            o[dt] = __builtin_amdgcn_wmma_f32_16x16x32_bf16(false, pa.v, false, vf.v,
                                                            (short)0, o[dt], false, false);
        }
    }

    // ---- normalize and write [S][NH*HD] f32
#pragma unroll
    for (int r = 0; r < 8; ++r) {
        float inv = (l_[r] > 0.0f) ? (1.0f / l_[r]) : 0.0f;
        int m = q0 + r + 8 * lh;
#pragma unroll
        for (int dt = 0; dt < 8; ++dt)
            Out[(size_t)m * (NH * HD) + h * HD + dt * 16 + ln] = o[dt][r] * inv;
    }
}

// =====================================================================
// launch
// =====================================================================
extern "C" void kernel_launch(void* const* d_in, const int* in_sizes, int n_in,
                              void* d_out, int out_size, void* d_ws, size_t ws_size,
                              hipStream_t stream)
{
    const float* hs   = (const float*)d_in[0];
    const int*   pos  = (const int*)d_in[1];
    const unsigned char* mask = (const unsigned char*)d_in[2];
    const float* wqkv = (const float*)d_in[3];
    const float* lqA  = (const float*)d_in[4];
    const float* lqB  = (const float*)d_in[5];
    const float* wo   = (const float*)d_in[6];
    const float* loA  = (const float*)d_in[7];
    const float* loB  = (const float*)d_in[8];
    float* out = (float*)d_out;

    char* ws = (char*)d_ws;
    // qkv (50.3 MB) is dead after rope_pack; attn output aliases it.
    float*  qkv  = (float*)ws;                       // S*QKV_OUT f32
    float*  attn = (float*)ws;                       // S*NH*HD  f32 (alias)
    size_t off = (size_t)SEQ * QKV_OUT * sizeof(float);
    float*  mid  = (float*)(ws + off);  off += (size_t)SEQ * LORA_R * sizeof(float);
    __bf16* Qb   = (__bf16*)(ws + off); off += (size_t)NH  * SEQ * HD * sizeof(__bf16);
    __bf16* Kb   = (__bf16*)(ws + off); off += (size_t)NKV * SEQ * HD * sizeof(__bf16);
    __bf16* Vt   = (__bf16*)(ws + off);

    dim3 blk(256);

    // 1) qkv = hs @ wqkv^T
    gemm_bf16_wmma<0><<<dim3(QKV_OUT / BN, SEQ / BM), blk, 0, stream>>>(
        hs, wqkv, qkv, SEQ, QKV_OUT, HID, nullptr);
    // 2) mid = hs @ loraA^T
    gemm_bf16_wmma<0><<<dim3(LORA_R / BN, SEQ / BM), blk, 0, stream>>>(
        hs, lqA, mid, SEQ, LORA_R, HID, nullptr);
    // 3) qkv += mask * (mid @ loraB^T)
    gemm_bf16_wmma<1><<<dim3(QKV_OUT / BN, SEQ / BM), blk, 0, stream>>>(
        mid, lqB, qkv, SEQ, QKV_OUT, LORA_R, mask);
    // 4) RoPE + repack to bf16
    rope_pack<<<(SEQ * NKV * 6 * 64) / 256, blk, 0, stream>>>(qkv, pos, Qb, Kb, Vt);
    // 5) flash attention (writes attn, aliasing dead qkv)
    attn_wmma<<<dim3(SEQ / (16 * 8), NH), blk, 0, stream>>>(Qb, Kb, Vt, attn);
    // 6) out = attn @ wo^T
    gemm_bf16_wmma<0><<<dim3(HID / BN, SEQ / BM), blk, 0, stream>>>(
        attn, wo, out, SEQ, HID, NH * HD, nullptr);
    // 7) mid = attn @ loA^T
    gemm_bf16_wmma<0><<<dim3(LORA_R / BN, SEQ / BM), blk, 0, stream>>>(
        attn, loA, mid, SEQ, LORA_R, NH * HD, nullptr);
    // 8) out += mask * (mid @ loB^T)
    gemm_bf16_wmma<1><<<dim3(HID / BN, SEQ / BM), blk, 0, stream>>>(
        mid, loB, out, SEQ, HID, LORA_R, mask);
}
